// GatedGraphConv_15616501088829
// MI455X (gfx1250) — matrix operand order repaired and checked by reading, not compiled, not run on hardware
//
#include <hip/hip_runtime.h>
#include <hip/hip_bf16.h>
#include <math.h>

// ---------------------------------------------------------------------------
// GatedGraphConv (DGL, n_steps=3, 1 edge type), N=50000 nodes, E=800000 edges,
// D=64. bf16 WMMA (v_wmma_f32_16x16x32_bf16) for all three GEMMs with f32
// accumulation; f32 atomicAdd scatter for segment_sum (L2-resident); fused
// GRU kernel (WMMA GEMMs -> LDS -> gates). Store guards are hoisted to one
// uniform (scalar-branch) full-tile test so the hot path has no EXEC churn.
// ---------------------------------------------------------------------------

#define D64 64
#define D192 192
#define LDSTR 200   // LDS row stride in floats (192 + 8 pad)

typedef __attribute__((ext_vector_type(16))) __bf16 v16bf;
typedef __attribute__((ext_vector_type(8)))  float  v8f;

// WMMA A operand (16x32 bf16) from row-major f32 X[.][64], converting inline.
// ISA 7.12.2 (16-bit A 16x32): lanes 0-15 hold K=0..7 & 16..23, lanes 16-31
// hold K=8..15 & 24..31; row M = lane&15.
__device__ __forceinline__ v16bf load_A_f32_as_bf16(const float* __restrict__ X,
                                                    int row, int k0, int laneHalf) {
    const float* p = X + row * D64 + k0 + 8 * laneHalf;
    v16bf a;
#pragma unroll
    for (int i = 0; i < 8; ++i) a[i]     = (__bf16)p[i];
#pragma unroll
    for (int i = 0; i < 8; ++i) a[8 + i] = (__bf16)p[16 + i];
    return a;
}

// WMMA B operand (32x16 bf16 tile of B = W^T) from row-major bf16 W[nOut][64]:
// column n = lane&15; lanes 0-15 hold K=0..7 & 16..23, lanes 16-31 K=8..15 & 24..31.
__device__ __forceinline__ v16bf load_B_from_W(const __bf16* __restrict__ W,
                                               int nRow, int k0, int laneHalf) {
    const __bf16* p = W + nRow * D64 + k0 + 8 * laneHalf;
    v16bf b;
#pragma unroll
    for (int i = 0; i < 8; ++i) b[i] = p[i];
#pragma unroll
    for (int i = 0; i < 8; ++i) b[8 + i] = p[16 + i];
    return b;
}

// ---------------------------------------------------------------------------
// One-time weight conversion f32 -> bf16
// ---------------------------------------------------------------------------
__global__ void ggc_convert_weights(const float* __restrict__ W,
                                    const float* __restrict__ Wih,
                                    const float* __restrict__ Whh,
                                    __bf16* __restrict__ Wbf,
                                    __bf16* __restrict__ Wihbf,
                                    __bf16* __restrict__ Whhbf) {
    int i = blockIdx.x * blockDim.x + threadIdx.x;
    if (i < D64 * D64) Wbf[i] = (__bf16)W[i];
    if (i < D192 * D64) {
        Wihbf[i] = (__bf16)Wih[i];
        Whhbf[i] = (__bf16)Whh[i];
    }
}

// ---------------------------------------------------------------------------
// wh = h @ W.T + b. Block = 128 threads (4 waves) owns 16 rows; wave w owns
// columns 16w..16w+15. K-loop: 2 x k32.
// ---------------------------------------------------------------------------
__global__ void ggc_wh_kernel(const float* __restrict__ h,
                              const __bf16* __restrict__ Wbf,
                              const float* __restrict__ bias,
                              float* __restrict__ wh, int nNodes) {
    const int wave = threadIdx.x >> 5;
    const int lane = threadIdx.x & 31;
    const int lh   = lane >> 4;
    const int ml   = lane & 15;
    const int m0   = blockIdx.x * 16;
    const int n0   = wave * 16;

    int arow = m0 + ml; if (arow >= nNodes) arow = nNodes - 1;  // clamp loads

    v8f c = {};
#pragma unroll
    for (int k0 = 0; k0 < D64; k0 += 32) {
        v16bf a = load_A_f32_as_bf16(h, arow, k0, lh);
        v16bf b = load_B_from_W(Wbf, n0 + ml, k0, lh);
        c = __builtin_amdgcn_wmma_f32_16x16x32_bf16(false, a, false, b,
                                                    (short)0, c, false, false);
    }
    const float bn = bias[n0 + ml];
    // D layout: row = 8*laneHalf + r, col = lane&15.
    if (m0 + 16 <= nNodes) {            // uniform (scalar) branch: fast path
#pragma unroll
        for (int r = 0; r < 8; ++r)
            wh[(m0 + 8 * lh + r) * D64 + n0 + ml] = c[r] + bn;
    } else {                            // remainder tile only
#pragma unroll
        for (int r = 0; r < 8; ++r) {
            int m = m0 + 8 * lh + r;
            if (m < nNodes) wh[m * D64 + n0 + ml] = c[r] + bn;
        }
    }
}

// ---------------------------------------------------------------------------
// a[] = 0
// ---------------------------------------------------------------------------
__global__ void ggc_zero_kernel(float* __restrict__ a, int n) {
    int i = blockIdx.x * blockDim.x + threadIdx.x;
    if (i < n) a[i] = 0.0f;
}

// ---------------------------------------------------------------------------
// a = segment_sum(wh[src], dst): 16 threads/edge, float4 gather + 4 f32
// atomics (global_atomic_add_f32; a[] is 12.8 MB -> L2-resident).
// ---------------------------------------------------------------------------
__global__ void ggc_scatter_kernel(const float* __restrict__ wh,
                                   const int* __restrict__ src,
                                   const int* __restrict__ dst,
                                   float* __restrict__ a, int nEdges) {
    int t = blockIdx.x * blockDim.x + threadIdx.x;
    int e = t >> 4;
    if (e >= nEdges) return;
    int l = t & 15;
    int s = src[e];
    int d = dst[e];
    float4 v = *((const float4*)(wh + s * D64) + l);
    float* q = a + d * D64 + l * 4;
    atomicAdd(q + 0, v.x);
    atomicAdd(q + 1, v.y);
    atomicAdd(q + 2, v.z);
    atomicAdd(q + 3, v.w);
}

// ---------------------------------------------------------------------------
// Fused GRU cell: gi = a@Wih.T + bih, gh = h@Whh.T + bhh (WMMA -> LDS), then
// r = sig(i_r+h_r), z = sig(i_z+h_z), n = tanh(i_n + r*h_n), h' = (1-z)n + z·h.
// Block = 128 threads (4 waves) owns 16 nodes; 24 WMMA tiles, 6 per wave.
// ---------------------------------------------------------------------------
__global__ void ggc_gru_kernel(const float* __restrict__ a,
                               const float* __restrict__ h,
                               const __bf16* __restrict__ Wih,
                               const __bf16* __restrict__ Whh,
                               const float* __restrict__ bih,
                               const float* __restrict__ bhh,
                               float* __restrict__ hout, int nNodes) {
    __shared__ float gi[16 * LDSTR];
    __shared__ float gh[16 * LDSTR];

    const int wave = threadIdx.x >> 5;
    const int lane = threadIdx.x & 31;
    const int lh   = lane >> 4;
    const int ml   = lane & 15;
    const int m0   = blockIdx.x * 16;

    int arow = m0 + ml; if (arow >= nNodes) arow = nNodes - 1;

    for (int t = wave; t < 24; t += 4) {           // uniform across waves
        const bool  isGh = (t >= 12);
        const int   n0   = (isGh ? t - 12 : t) * 16;
        const float*  X  = isGh ? h : a;
        const __bf16* W  = isGh ? Whh : Wih;
        const float*  bb = isGh ? bhh : bih;
        float*        out = isGh ? gh : gi;

        v8f c = {};
#pragma unroll
        for (int k0 = 0; k0 < D64; k0 += 32) {
            v16bf av = load_A_f32_as_bf16(X, arow, k0, lh);
            v16bf bv = load_B_from_W(W, n0 + ml, k0, lh);
            c = __builtin_amdgcn_wmma_f32_16x16x32_bf16(false, av, false, bv,
                                                        (short)0, c, false, false);
        }
        const float bn = bb[n0 + ml];
#pragma unroll
        for (int r = 0; r < 8; ++r)
            out[(8 * lh + r) * LDSTR + n0 + ml] = c[r] + bn;
    }
    __syncthreads();

    // Gate math: 16 nodes x 64 cols = 1024 outputs over 128 threads (8 each).
    const int m    = threadIdx.x >> 3;          // node within tile
    const int cb   = (threadIdx.x & 7) * 8;     // column base
    const int node = m0 + m;
    if (m0 + 16 <= nNodes) {                    // uniform fast path
#pragma unroll
        for (int j = 0; j < 8; ++j) {
            int cc = cb + j;
            float ir = gi[m * LDSTR + cc];
            float iz = gi[m * LDSTR + 64 + cc];
            float in = gi[m * LDSTR + 128 + cc];
            float hr = gh[m * LDSTR + cc];
            float hz = gh[m * LDSTR + 64 + cc];
            float hn = gh[m * LDSTR + 128 + cc];
            float r = 1.0f / (1.0f + __expf(-(ir + hr)));
            float z = 1.0f / (1.0f + __expf(-(iz + hz)));
            float n = tanhf(in + r * hn);
            float ho = h[node * D64 + cc];
            hout[node * D64 + cc] = (1.0f - z) * n + z * ho;
        }
    } else if (node < nNodes) {                 // remainder tile
#pragma unroll
        for (int j = 0; j < 8; ++j) {
            int cc = cb + j;
            float ir = gi[m * LDSTR + cc];
            float iz = gi[m * LDSTR + 64 + cc];
            float in = gi[m * LDSTR + 128 + cc];
            float hr = gh[m * LDSTR + cc];
            float hz = gh[m * LDSTR + 64 + cc];
            float hn = gh[m * LDSTR + 128 + cc];
            float r = 1.0f / (1.0f + __expf(-(ir + hr)));
            float z = 1.0f / (1.0f + __expf(-(iz + hz)));
            float n = tanhf(in + r * hn);
            float ho = h[node * D64 + cc];
            hout[node * D64 + cc] = (1.0f - z) * n + z * ho;
        }
    }
}

// ---------------------------------------------------------------------------
// Launcher. Inputs (setup_inputs order):
//  0 node_in [N,64] f32 | 1 W [64,64] | 2 b [64] | 3 Wih [192,64] | 4 Whh [192,64]
//  5 bih [192] | 6 bhh [192] | 7 src [E] i32 | 8 dst [E] i32
// Workspace: wh, a, hA, hB (N*64 f32 each) + bf16 weights (~56 KB).
// ---------------------------------------------------------------------------
extern "C" void kernel_launch(void* const* d_in, const int* in_sizes, int n_in,
                              void* d_out, int out_size, void* d_ws, size_t ws_size,
                              hipStream_t stream) {
    const float* node_in = (const float*)d_in[0];
    const float* W       = (const float*)d_in[1];
    const float* b       = (const float*)d_in[2];
    const float* Wih     = (const float*)d_in[3];
    const float* Whh     = (const float*)d_in[4];
    const float* bih     = (const float*)d_in[5];
    const float* bhh     = (const float*)d_in[6];
    const int*   src     = (const int*)d_in[7];
    const int*   dst     = (const int*)d_in[8];

    const int nNodes = in_sizes[0] / D64;
    const int nEdges = in_sizes[7];

    float* wh = (float*)d_ws;
    float* a  = wh + (size_t)nNodes * D64;
    float* hA = a  + (size_t)nNodes * D64;
    float* hB = hA + (size_t)nNodes * D64;
    __bf16* Wbf   = (__bf16*)(hB + (size_t)nNodes * D64);
    __bf16* Wihbf = Wbf + D64 * D64;
    __bf16* Whhbf = Wihbf + D192 * D64;

    const int rowTiles   = (nNodes + 15) / 16;
    const int zeroBlocks = (nNodes * D64 + 255) / 256;
    const int scatBlocks = (nEdges * 16 + 255) / 256;

    ggc_convert_weights<<<(D192 * D64 + 255) / 256, 256, 0, stream>>>(
        W, Wih, Whh, Wbf, Wihbf, Whhbf);

    for (int step = 0; step < 3; ++step) {
        const float* h_in  = (step == 0) ? node_in : ((step == 1) ? hA : hB);
        float*       h_out = (step == 0) ? hA : ((step == 1) ? hB : (float*)d_out);

        ggc_wh_kernel<<<rowTiles, 128, 0, stream>>>(h_in, Wbf, b, wh, nNodes);
        ggc_zero_kernel<<<zeroBlocks, 256, 0, stream>>>(a, nNodes * D64);
        ggc_scatter_kernel<<<scatBlocks, 256, 0, stream>>>(wh, src, dst, a, nEdges);
        ggc_gru_kernel<<<rowTiles, 128, 0, stream>>>(a, h_in, Wihbf, Whhbf,
                                                     bih, bhh, h_out, nNodes);
    }
}
